// RnnSenderFixedLengthGS_88639535055420
// MI455X (gfx1250) — matrix operand order repaired and checked by reading, not compiled, not run on hardware
//
#include <hip/hip_runtime.h>

// ---------------------------------------------------------------------------
// RnnSenderFixedLengthGS for MI455X (gfx1250, wave32, WMMA bf16 + async LDS)
// B=128, F=2048, H=1024, E=512, V=16384, T=20, TEMP=1
// ---------------------------------------------------------------------------

#define DI __device__ __forceinline__

#if defined(__has_builtin)
#if __has_builtin(__builtin_amdgcn_global_load_async_to_lds_b128)
#define HAVE_ASYNC_COPY 1
#endif
#if __has_builtin(__builtin_amdgcn_s_wait_asynccnt)
#define HAVE_WAIT_ASYNC 1
#endif
#endif

namespace cfg {
constexpr int B = 128, F = 2048, H = 1024, E = 512, V = 16384, T = 20;
}

typedef __bf16 bf16_t;
typedef bf16_t bf16x16 __attribute__((ext_vector_type(16)));
typedef float  f32x8   __attribute__((ext_vector_type(8)));
typedef int    v4i     __attribute__((vector_size(4 * sizeof(int))));

union Frag {
  bf16x16 v;
  uint4   q[2];
};

DI f32x8 zero8() {
  f32x8 z;
#pragma unroll
  for (int i = 0; i < 8; ++i) z[i] = 0.0f;
  return z;
}

// D = A(16x32 bf16) x B(32x16 bf16) + C(16x16 f32), wave32
DI f32x8 wmma_acc(f32x8 c, const Frag& a, const Frag& b) {
  return __builtin_amdgcn_wmma_f32_16x16x32_bf16(
      /*neg_a=*/false, a.v, /*neg_b=*/false, b.v,
      /*c_mod=*/(short)0, c, /*reuse_a=*/false, /*reuse_b=*/false);
}

// A-operand: row-major [.., K] bf16.  lane<16 -> M=lane, K chunks
// {klo..klo+7} in v0..v3 and {klo+16..klo+23} in v4..v7, klo=(lane>>4)*8.
DI Frag load_a_frag(const unsigned short* __restrict__ A, int mbase, int K,
                    int kb, int lane) {
  const int m   = lane & 15;
  const int klo = (lane >> 4) << 3;
  const unsigned short* p = A + (size_t)(mbase + m) * K + (kb + klo);
  Frag f;
  f.q[0] = *(const uint4*)(p);
  f.q[1] = *(const uint4*)(p + 16);
  return f;
}

// B-operand from global: weights pre-transposed K-major [K][N] bf16.
// lane -> N column (lane&15), K rows {klo..klo+7, klo+16..klo+23}, klo=(lane>>4)*8.
DI Frag load_b_frag(const unsigned short* __restrict__ Bt, int N, int kb,
                    int n0, int lane) {
  const int klo = (lane >> 4) << 3;
  const unsigned short* p = Bt + (size_t)(kb + klo) * N + n0 + (lane & 15);
  Frag f;
#pragma unroll
  for (int i = 0; i < 8; ++i) {
    f.v[i]     = __builtin_bit_cast(bf16_t, p[(size_t)i * N]);
    f.v[8 + i] = __builtin_bit_cast(bf16_t, p[(size_t)(16 + i) * N]);
  }
  return f;
}

// B-operand from an LDS tile laid out [32][W] bf16 (row = K).
DI Frag load_b_frag_lds(const unsigned short* base, int W, int n0, int lane) {
  const int klo = (lane >> 4) << 3;
  const unsigned short* p = base + klo * W + n0 + (lane & 15);
  Frag f;
#pragma unroll
  for (int i = 0; i < 8; ++i) {
    f.v[i]     = __builtin_bit_cast(bf16_t, p[i * W]);
    f.v[8 + i] = __builtin_bit_cast(bf16_t, p[(16 + i) * W]);
  }
  return f;
}

// stage 16 bytes global -> LDS (async if available)
DI void copy16_g2l(const unsigned short* gp, unsigned short* lp) {
#if defined(HAVE_ASYNC_COPY)
  __builtin_amdgcn_global_load_async_to_lds_b128((v4i*)gp, (v4i*)lp, 0, 0);
#else
  *(uint4*)lp = *(const uint4*)gp;
#endif
}

DI void wait_async_copies() {
#if defined(HAVE_ASYNC_COPY)
#if defined(HAVE_WAIT_ASYNC)
  __builtin_amdgcn_s_wait_asynccnt(0);
#else
  asm volatile("s_wait_asynccnt 0" ::: "memory");
#endif
#endif
}

DI unsigned short f32_to_bf16(float f) {
  unsigned int u = __float_as_uint(f);
  u += 0x7FFFu + ((u >> 16) & 1u);  // round-to-nearest-even
  return (unsigned short)(u >> 16);
}

DI float sigmoidf_(float x) { return 1.0f / (1.0f + __expf(-x)); }

DI void atomic_max_float(float* addr, float v) {
  int iv = __float_as_int(v);
  if (iv >= 0)
    atomicMax((int*)addr, iv);
  else
    atomicMin((unsigned int*)addr, (unsigned int)iv);
}

// ---------------------------------------------------------------------------
// Conversion / init kernels
// ---------------------------------------------------------------------------

// dst[k*N + n] = bf16(src[n*K + k])   (src is [N][K] row-major)
__global__ void k_transpose_bf16(const float* __restrict__ src,
                                 unsigned short* __restrict__ dst, int N, int K) {
  size_t idx = (size_t)blockIdx.x * blockDim.x + threadIdx.x;
  if (idx >= (size_t)N * K) return;
  int n = (int)(idx / K);
  int k = (int)(idx % K);
  dst[(size_t)k * N + n] = f32_to_bf16(src[idx]);
}

__global__ void k_convert_bf16(const float* __restrict__ src,
                               unsigned short* __restrict__ dst, int total) {
  int idx = blockIdx.x * blockDim.x + threadIdx.x;
  if (idx < total) dst[idx] = f32_to_bf16(src[idx]);
}

__global__ void k_fill_sos(const float* __restrict__ sos,
                           unsigned short* __restrict__ eb) {
  int idx = blockIdx.x * blockDim.x + threadIdx.x;
  if (idx < cfg::B * cfg::E) eb[idx] = f32_to_bf16(sos[idx % cfg::E]);
}

__global__ void k_init_rowmax(float* __restrict__ rowmax) {
  int i = threadIdx.x;
  if (i < cfg::B) rowmax[i] = __int_as_float(0xff800000);  // -inf
}

// ---------------------------------------------------------------------------
// h0 = image_features @ agent_w.T + agent_b   (M=128, K=2048, N=1024)
// grid = H/16, block = 256 (8 waves, one 16-row M block each). One-time cost.
// ---------------------------------------------------------------------------
__global__ __launch_bounds__(256) void k_h0(
    const unsigned short* __restrict__ imgb,    // [B,F] bf16
    const unsigned short* __restrict__ agent_t, // [F,H] bf16 (K-major)
    const float* __restrict__ agent_b,
    float* __restrict__ hf, unsigned short* __restrict__ hb) {
  using namespace cfg;
  const int lane = threadIdx.x & 31, wave = threadIdx.x >> 5;
  const int mrow = wave * 16;
  const int n0 = blockIdx.x * 16;
  f32x8 acc = zero8();
  for (int kb = 0; kb < F; kb += 32) {
    Frag a = load_a_frag(imgb, mrow, F, kb, lane);
    Frag b = load_b_frag(agent_t, H, kb, n0, lane);
    acc = wmma_acc(acc, a, b);
  }
  const int nloc = lane & 15, half = lane >> 4;
  const int col = n0 + nloc;
  const float bo = agent_b[col];
#pragma unroll
  for (int r = 0; r < 8; ++r) {
    int row = mrow + r + 8 * half;
    float v = acc[r] + bo;
    hf[(size_t)row * H + col] = v;
    hb[(size_t)row * H + col] = f32_to_bf16(v);
  }
}

// ---------------------------------------------------------------------------
// Fused GRU cell: h' = (1-z)*n + z*h    grid = H/16, block = 256.
// B tiles (3 gates x 32K x 16N) double-buffered in LDS via async copies:
// removes the 8x cross-wave redundancy of the weight reads.
// ---------------------------------------------------------------------------
__global__ __launch_bounds__(256) void k_gru(
    const unsigned short* __restrict__ eb,     // [B,E]  bf16  (x input)
    const unsigned short* __restrict__ hinb,   // [B,H]  bf16
    const float* __restrict__ hinf,            // [B,H]  f32
    const unsigned short* __restrict__ wih_t,  // [E,3H] bf16
    const unsigned short* __restrict__ whh_t,  // [H,3H] bf16
    const float* __restrict__ b_ih, const float* __restrict__ b_hh,
    float* __restrict__ houtf, unsigned short* __restrict__ houtb) {
  using namespace cfg;
  const int N3 = 3 * H;
  const int tid = threadIdx.x;
  const int lane = tid & 31, wave = tid >> 5;
  const int mrow = wave * 16;
  const int j = blockIdx.x * 16;

  // [buf][gate][32 K-rows][16 N] bf16  -> 2 * 3KB
  __shared__ __align__(16) unsigned short sB[2][3 * 32 * 16];

  // staging: 192 threads move 3 * (32x16) bf16 = 3KB per k-step (16B each)
  const int sg    = tid / 64;            // gate 0..2 (valid for tid<192)
  const int srow  = (tid % 64) >> 1;     // 0..31
  const int sseg  = (tid & 1) * 8;       // 0 or 8

  f32x8 air = zero8(), aiz = zero8(), ain = zero8();
  f32x8 ahr = zero8(), ahz = zero8(), ahn = zero8();

  // -------- phase 1: input gates over K=E --------
  if (tid < 192)
    copy16_g2l(wih_t + (size_t)srow * N3 + sg * H + j + sseg,
               &sB[0][(sg * 32 + srow) * 16 + sseg]);
  for (int kb = 0, it = 0; kb < E; kb += 32, ++it) {
    const int cur = it & 1;
    wait_async_copies();
    __syncthreads();
    if (kb + 32 < E && tid < 192)
      copy16_g2l(wih_t + (size_t)(kb + 32 + srow) * N3 + sg * H + j + sseg,
                 &sB[cur ^ 1][(sg * 32 + srow) * 16 + sseg]);
    Frag a = load_a_frag(eb, mrow, E, kb, lane);
    air = wmma_acc(air, a, load_b_frag_lds(&sB[cur][0 * 512], 16, 0, lane));
    aiz = wmma_acc(aiz, a, load_b_frag_lds(&sB[cur][1 * 512], 16, 0, lane));
    ain = wmma_acc(ain, a, load_b_frag_lds(&sB[cur][2 * 512], 16, 0, lane));
  }
  __syncthreads();

  // -------- phase 2: hidden gates over K=H --------
  if (tid < 192)
    copy16_g2l(whh_t + (size_t)srow * N3 + sg * H + j + sseg,
               &sB[0][(sg * 32 + srow) * 16 + sseg]);
  for (int kb = 0, it = 0; kb < H; kb += 32, ++it) {
    const int cur = it & 1;
    wait_async_copies();
    __syncthreads();
    if (kb + 32 < H && tid < 192)
      copy16_g2l(whh_t + (size_t)(kb + 32 + srow) * N3 + sg * H + j + sseg,
                 &sB[cur ^ 1][(sg * 32 + srow) * 16 + sseg]);
    Frag a = load_a_frag(hinb, mrow, H, kb, lane);
    ahr = wmma_acc(ahr, a, load_b_frag_lds(&sB[cur][0 * 512], 16, 0, lane));
    ahz = wmma_acc(ahz, a, load_b_frag_lds(&sB[cur][1 * 512], 16, 0, lane));
    ahn = wmma_acc(ahn, a, load_b_frag_lds(&sB[cur][2 * 512], 16, 0, lane));
  }

  const int nloc = lane & 15, half = lane >> 4;
  const int col = j + nloc;
  const float bir = b_ih[col],         bhr = b_hh[col];
  const float biz = b_ih[H + col],     bhz = b_hh[H + col];
  const float bin = b_ih[2 * H + col], bhn = b_hh[2 * H + col];
#pragma unroll
  for (int r = 0; r < 8; ++r) {
    int row = mrow + r + 8 * half;
    float rg = sigmoidf_(air[r] + bir + ahr[r] + bhr);
    float zg = sigmoidf_(aiz[r] + biz + ahz[r] + bhz);
    float ng = tanhf(ain[r] + bin + rg * (ahn[r] + bhn));
    float hp = hinf[(size_t)row * H + col];
    float hv = (1.0f - zg) * ng + zg * hp;
    houtf[(size_t)row * H + col] = hv;
    houtb[(size_t)row * H + col] = f32_to_bf16(hv);
  }
}

// ---------------------------------------------------------------------------
// logits = h @ w_out.T + b_out + gumbel_t ; per-row max via atomics.
// grid = V/64, block = 256. 32x64 B tile double-buffered in LDS (async).
// ---------------------------------------------------------------------------
__global__ __launch_bounds__(256) void k_logits(
    const unsigned short* __restrict__ hb,     // [B,H] bf16
    const unsigned short* __restrict__ wout_t, // [H,V] bf16
    const float* __restrict__ b_out,
    const float* __restrict__ gum,             // [B,V] (this step)
    float* __restrict__ logits,                // [B,V] scratch
    float* __restrict__ rowmax) {
  using namespace cfg;
  const int tid = threadIdx.x;
  const int lane = tid & 31, wave = tid >> 5;
  const int mrow = wave * 16;
  const int n0 = blockIdx.x * 64;

  __shared__ __align__(16) unsigned short sB[2][32 * 64];  // 2 x 4KB

  const int skrow = tid >> 3;       // 0..31
  const int sseg  = (tid & 7) * 8;  // 0..56

  f32x8 acc[4] = {zero8(), zero8(), zero8(), zero8()};

  copy16_g2l(wout_t + (size_t)skrow * V + n0 + sseg,
             &sB[0][skrow * 64 + sseg]);
  for (int kb = 0, it = 0; kb < H; kb += 32, ++it) {
    const int cur = it & 1;
    wait_async_copies();
    __syncthreads();
    if (kb + 32 < H)
      copy16_g2l(wout_t + (size_t)(kb + 32 + skrow) * V + n0 + sseg,
                 &sB[cur ^ 1][skrow * 64 + sseg]);
    Frag a = load_a_frag(hb, mrow, H, kb, lane);
#pragma unroll
    for (int s = 0; s < 4; ++s)
      acc[s] = wmma_acc(acc[s], a, load_b_frag_lds(&sB[cur][0], 64, 16 * s, lane));
  }

  const int nloc = lane & 15, half = lane >> 4;
  float rmax[8];
#pragma unroll
  for (int r = 0; r < 8; ++r) rmax[r] = __int_as_float(0xff800000);

#pragma unroll
  for (int s = 0; s < 4; ++s) {
    int col = n0 + 16 * s + nloc;
    float bo = b_out[col];
#pragma unroll
    for (int r = 0; r < 8; ++r) {
      int row = mrow + r + 8 * half;
      float l = acc[s][r] + bo + gum[(size_t)row * V + col];
      logits[(size_t)row * V + col] = l;
      rmax[r] = fmaxf(rmax[r], l);
    }
  }
#pragma unroll
  for (int r = 0; r < 8; ++r) {
    float v = rmax[r];
    v = fmaxf(v, __shfl_xor(v, 1, 32));
    v = fmaxf(v, __shfl_xor(v, 2, 32));
    v = fmaxf(v, __shfl_xor(v, 4, 32));
    v = fmaxf(v, __shfl_xor(v, 8, 32));
    if (nloc == 0) atomic_max_float(&rowmax[mrow + r + 8 * half], v);
  }
}

// ---------------------------------------------------------------------------
// softmax (TEMP=1): x = exp(l - max)/sum ; fp32 to output + bf16 copy
// grid = B, block = 256
// ---------------------------------------------------------------------------
__global__ __launch_bounds__(256) void k_softmax(
    float* __restrict__ logits, const float* __restrict__ rowmax,
    float* __restrict__ out, int t, unsigned short* __restrict__ xb) {
  using namespace cfg;
  const int b = blockIdx.x, tid = threadIdx.x;
  __shared__ float sred[256];
  __shared__ float sinv;
  const float m = rowmax[b];
  float lsum = 0.0f;
  for (int v = tid; v < V; v += 256) {
    float ex = __expf(logits[(size_t)b * V + v] - m);
    logits[(size_t)b * V + v] = ex;
    lsum += ex;
  }
  sred[tid] = lsum;
  __syncthreads();
  for (int s = 128; s > 0; s >>= 1) {
    if (tid < s) sred[tid] += sred[tid + s];
    __syncthreads();
  }
  if (tid == 0) sinv = 1.0f / sred[0];
  __syncthreads();
  const float inv = sinv;
  float* orow = out + ((size_t)b * T + t) * V;
  for (int v = tid; v < V; v += 256) {
    float x = logits[(size_t)b * V + v] * inv;
    orow[v] = x;
    xb[(size_t)b * V + v] = f32_to_bf16(x);
  }
}

// ---------------------------------------------------------------------------
// e = x @ emb_w.T   (M=128, K=16384, N=512) -> bf16 for next GRU step
// grid = E/32, block = 256 (2 subtiles per wave); A-traffic dominates here.
// ---------------------------------------------------------------------------
__global__ __launch_bounds__(256) void k_emb(
    const unsigned short* __restrict__ xb,    // [B,V] bf16
    const unsigned short* __restrict__ emb_t, // [V,E] bf16
    unsigned short* __restrict__ eb) {        // [B,E] bf16
  using namespace cfg;
  const int lane = threadIdx.x & 31, wave = threadIdx.x >> 5;
  const int mrow = wave * 16;
  const int n0 = blockIdx.x * 32;
  f32x8 acc[2] = {zero8(), zero8()};
  for (int kb = 0; kb < V; kb += 32) {
    Frag a = load_a_frag(xb, mrow, V, kb, lane);
#pragma unroll
    for (int s = 0; s < 2; ++s)
      acc[s] = wmma_acc(acc[s], a, load_b_frag(emb_t, E, kb, n0 + 16 * s, lane));
  }
  const int nloc = lane & 15, half = lane >> 4;
#pragma unroll
  for (int s = 0; s < 2; ++s) {
    int col = n0 + 16 * s + nloc;
#pragma unroll
    for (int r = 0; r < 8; ++r) {
      int row = mrow + r + 8 * half;
      eb[(size_t)row * E + col] = f32_to_bf16(acc[s][r]);
    }
  }
}

// ---------------------------------------------------------------------------
// Host launcher
// ---------------------------------------------------------------------------
extern "C" void kernel_launch(void* const* d_in, const int* in_sizes, int n_in,
                              void* d_out, int out_size, void* d_ws,
                              size_t ws_size, hipStream_t stream) {
  using namespace cfg;
  const float* img      = (const float*)d_in[0];
  const float* agent_w  = (const float*)d_in[1];
  const float* agent_b  = (const float*)d_in[2];
  const float* w_ih     = (const float*)d_in[3];
  const float* w_hh     = (const float*)d_in[4];
  const float* b_ih     = (const float*)d_in[5];
  const float* b_hh     = (const float*)d_in[6];
  const float* w_out    = (const float*)d_in[7];
  const float* b_out    = (const float*)d_in[8];
  const float* emb_w    = (const float*)d_in[9];
  const float* sos      = (const float*)d_in[10];
  const float* gumbel   = (const float*)d_in[11];
  float* out = (float*)d_out;

  char* wp = (char*)d_ws;
  auto alloc = [&](size_t bytes) -> void* {
    void* p = (void*)wp;
    wp += (bytes + 255) & ~(size_t)255;
    return p;
  };
  unsigned short* agent_t = (unsigned short*)alloc((size_t)F * H * 2);
  unsigned short* wih_t   = (unsigned short*)alloc((size_t)E * 3 * H * 2);
  unsigned short* whh_t   = (unsigned short*)alloc((size_t)H * 3 * H * 2);
  unsigned short* wout_t  = (unsigned short*)alloc((size_t)H * V * 2);
  unsigned short* emb_t   = (unsigned short*)alloc((size_t)V * E * 2);
  unsigned short* imgb    = (unsigned short*)alloc((size_t)B * F * 2);
  unsigned short* eb      = (unsigned short*)alloc((size_t)B * E * 2);
  float*          hAf     = (float*)alloc((size_t)B * H * 4);
  float*          hBf     = (float*)alloc((size_t)B * H * 4);
  unsigned short* hAb     = (unsigned short*)alloc((size_t)B * H * 2);
  unsigned short* hBb     = (unsigned short*)alloc((size_t)B * H * 2);
  float*          logits  = (float*)alloc((size_t)B * V * 4);
  unsigned short* xb      = (unsigned short*)alloc((size_t)B * V * 2);
  float*          rowmax  = (float*)alloc((size_t)B * 4);

  auto g1 = [](size_t n) { return (unsigned)((n + 255) / 256); };

  // per-launch weight conversion into WMMA-friendly K-major bf16
  k_transpose_bf16<<<g1((size_t)H * F), 256, 0, stream>>>(agent_w, agent_t, H, F);
  k_transpose_bf16<<<g1((size_t)3 * H * E), 256, 0, stream>>>(w_ih, wih_t, 3 * H, E);
  k_transpose_bf16<<<g1((size_t)3 * H * H), 256, 0, stream>>>(w_hh, whh_t, 3 * H, H);
  k_transpose_bf16<<<g1((size_t)V * H), 256, 0, stream>>>(w_out, wout_t, V, H);
  k_transpose_bf16<<<g1((size_t)E * V), 256, 0, stream>>>(emb_w, emb_t, E, V);
  k_convert_bf16<<<g1((size_t)B * F), 256, 0, stream>>>(img, imgb, B * F);
  k_fill_sos<<<g1((size_t)B * E), 256, 0, stream>>>(sos, eb);

  k_h0<<<H / 16, 256, 0, stream>>>(imgb, agent_t, agent_b, hAf, hAb);

  for (int t = 0; t < T; ++t) {
    const bool even = (t & 1) == 0;
    const float*          hinf  = even ? hAf : hBf;
    const unsigned short* hinb  = even ? hAb : hBb;
    float*                houtf = even ? hBf : hAf;
    unsigned short*       houtb = even ? hBb : hAb;

    k_gru<<<H / 16, 256, 0, stream>>>(eb, hinb, hinf, wih_t, whh_t, b_ih, b_hh,
                                      houtf, houtb);
    k_init_rowmax<<<1, 128, 0, stream>>>(rowmax);
    k_logits<<<V / 64, 256, 0, stream>>>(houtb, wout_t, b_out,
                                         gumbel + (size_t)t * B * V, logits,
                                         rowmax);
    k_softmax<<<B, 256, 0, stream>>>(logits, rowmax, out, t, xb);
    k_emb<<<E / 32, 256, 0, stream>>>(xb, emb_t, eb);
  }
}
